// RIGNO_SR_71949292142784
// MI455X (gfx1250) — matrix operand order repaired and checked by reading, not compiled, not run on hardware
//
#include <hip/hip_runtime.h>
#include <hip/hip_bf16.h>

// RIGNO GNN forward on gfx1250 (MI455X), wave32 + WMMA f16 (f32 accumulate).
// All FFNs are computed as 16-row tiles per wave with v_wmma_f32_16x16x32_f16;
// weights are pre-swizzled into WMMA B-fragment layout once per call.
// Params dict assumed flattened in JAX pytree order (sorted keys):
// outer: dec_ee,dec_no,dec_pe,dec_pn,enc_ee,enc_en,enc_no,enc_pe,enc_pn,
//        proc_e0..3,proc_ee,proc_n0..3,proc_no ; inner: W1,W2,b1,b2[,beta,g]

typedef __attribute__((ext_vector_type(16))) _Float16 v16h;
typedef __attribute__((ext_vector_type(8)))  _Float16 v8h;
typedef __attribute__((ext_vector_type(8)))  float    v8f;

#define WPB  4     // waves per block
#define TR   16    // rows per wave tile
#define MAXK 192   // max padded K (concat of 3x64)

// A fragment (16x32 f16) from LDS row-major tile, per CDNA5 ISA layout:
// lanes 0-15: row=lane, halves 0-7 -> K 0..7,  halves 8-15 -> K 16..23
// lanes16-31: row=lane-16, halves 0-7 -> K 8..15, halves 8-15 -> K 24..31
__device__ __forceinline__ v16h load_a_frag(const _Float16* X, int ld, int kt, int lane) {
  const _Float16* p = X + (lane & 15) * ld + kt * 32 + (lane >> 4) * 8;
  v8h lo = *(const v8h*)(p);
  v8h hi = *(const v8h*)(p + 16);
  v16h a;
#pragma unroll
  for (int i = 0; i < 8; ++i) { a[i] = lo[i]; a[i + 8] = hi[i]; }
  return a;
}

// B fragment: pre-swizzled in global memory, 32 bytes contiguous per lane.
__device__ __forceinline__ v16h load_b_frag(const _Float16* q) {
  v8h lo = *(const v8h*)(q);
  v8h hi = *(const v8h*)(q + 8);
  v16h b;
#pragma unroll
  for (int i = 0; i < 8; ++i) { b[i] = lo[i]; b[i + 8] = hi[i]; }
  return b;
}

// Generic fused FFN: gather/concat -> GEMM1 -> act -> GEMM2 -> (LN) ->
// (residual) -> (store) -> (ef += y) -> (atomic scatter-add + count).
__global__ __launch_bounds__(WPB * 32) void ffn_wmma_kernel(
    const float* __restrict__ srcA, const int* __restrict__ idxA, int dinA,
    const float* __restrict__ srcB, const int* __restrict__ idxB,
    const float* __restrict__ cntB, int dinB,
    const float* __restrict__ srcC, int dinC,
    const _Float16* __restrict__ W1f, int kt1, const float* __restrict__ b1,
    const _Float16* __restrict__ W2f, int nt2, const float* __restrict__ b2,
    int outCols,
    const float* __restrict__ lnG, const float* __restrict__ lnB,
    int act,                               // 0 = silu, 1 = sigmoid (hidden)
    const float* __restrict__ resid,
    float* __restrict__ outPtr, int ldOut,
    float* __restrict__ efPtr,             // if set: efPtr[row] += y
    float* __restrict__ scatSum, float* __restrict__ scatCnt,
    const int* __restrict__ scatIdx,
    int nRows)
{
  __shared__ _Float16 sX[WPB][TR * MAXK];
  __shared__ float    sY[WPB][TR * 64];
  const int wave = threadIdx.x >> 5;
  const int lane = threadIdx.x & 31;
  const int row0 = (blockIdx.x * WPB + wave) * TR;
  _Float16* X = sX[wave];
  float*    Y = sY[wave];
  const int Kpad = kt1 * 32;
  const int dAB  = dinA + dinB;
  const int dTot = dAB + dinC;

  // ---- gather + convert to f16, zero pad to Kpad ----
  for (int e = lane; e < TR * Kpad; e += 32) {
    int r  = e / Kpad;
    int c  = e - r * Kpad;
    int gr = row0 + r;
    float v = 0.0f;
    if (gr < nRows && c < dTot) {
      if (c < dinA) {
        int ra = idxA ? idxA[gr * 2] : gr;
        v = srcA[(long)ra * dinA + c];
      } else if (c < dAB) {
        int rb = idxB ? idxB[gr * 2] : gr;
        v = srcB[(long)rb * dinB + (c - dinA)];
        if (cntB) v /= fmaxf(cntB[rb], 1.0f);     // scatter-mean divide
      } else {
        v = srcC[(long)gr * dinC + (c - dAB)];
      }
    }
    X[r * Kpad + c] = (_Float16)v;
  }
  __syncthreads();

  // ---- GEMM1: (16 x Kpad) @ (Kpad x 64) -> Y f32 ----
  {
    const int g  = lane >> 4;
    const int cl = lane & 15;
    for (int nt = 0; nt < 4; ++nt) {
      v8f c = {};
      for (int kt = 0; kt < kt1; ++kt) {
        v16h a = load_a_frag(X, Kpad, kt, lane);
        v16h b = load_b_frag(W1f + ((kt * 4 + nt) * 32 + lane) * 16);
        c = __builtin_amdgcn_wmma_f32_16x16x32_f16(false, a, false, b,
                                                   (short)0, c, false, false);
      }
#pragma unroll
      for (int j = 0; j < 8; ++j)
        Y[(j + 8 * g) * 64 + nt * 16 + cl] = c[j];
    }
  }
  __syncthreads();

  // ---- hidden bias + activation -> X reused as 16x64 f16 ----
  for (int e = lane; e < TR * 64; e += 32) {
    float h = Y[e] + b1[e & 63];
    float s = 1.0f / (1.0f + __expf(-h));
    h = act ? s : h * s;                    // sigmoid vs silu
    X[e] = (_Float16)h;
  }
  __syncthreads();

  // ---- GEMM2: (16 x 64) @ (64 x out) -> Y f32 (+b2) ----
  {
    const int g  = lane >> 4;
    const int cl = lane & 15;
    for (int nt = 0; nt < nt2; ++nt) {
      v8f c = {};
#pragma unroll
      for (int kt = 0; kt < 2; ++kt) {
        v16h a = load_a_frag(X, 64, kt, lane);
        v16h b = load_b_frag(W2f + ((kt * nt2 + nt) * 32 + lane) * 16);
        c = __builtin_amdgcn_wmma_f32_16x16x32_f16(false, a, false, b,
                                                   (short)0, c, false, false);
      }
      int col  = nt * 16 + cl;
      float bb = (col < outCols) ? b2[col] : 0.0f;
#pragma unroll
      for (int j = 0; j < 8; ++j)
        Y[(j + 8 * g) * 64 + col] = c[j] + bb;
    }
  }
  __syncthreads();

  // ---- LayerNorm over 64 cols (rows owned by lanes 0..15) ----
  if (lnG) {
    if (lane < 16) {
      float* yr = Y + lane * 64;
      float mu = 0.0f;
      for (int c = 0; c < 64; ++c) mu += yr[c];
      mu *= (1.0f / 64.0f);
      float var = 0.0f;
      for (int c = 0; c < 64; ++c) { float d = yr[c] - mu; var += d * d; }
      var *= (1.0f / 64.0f);
      float rs = rsqrtf(var + 1e-5f);
      for (int c = 0; c < 64; ++c)
        yr[c] = (yr[c] - mu) * rs * lnG[c] + lnB[c];
    }
    __syncthreads();
  }

  // ---- epilogue: residual / store / ef update / scatter-add ----
  for (int e = lane; e < TR * outCols; e += 32) {
    int r  = e / outCols;
    int c  = e - r * outCols;
    int gr = row0 + r;
    if (gr >= nRows) continue;
    float y = Y[r * 64 + c];
    if (resid)  y += resid[(long)gr * 64 + c];
    if (outPtr) outPtr[(long)gr * ldOut + c] = y;
    if (efPtr)  efPtr[(long)gr * 64 + c] += y;
    if (scatSum) {
      int tr = scatIdx[gr * 2];
      atomicAdd(&scatSum[(long)tr * 64 + c], y);
      if (c == 0) atomicAdd(&scatCnt[tr], 1.0f);
    }
  }
}

// Pre-swizzle f32 weights (rows x cols row-major) into WMMA B fragments:
// half i of lane l of tile (kt,nt) holds W[kt*32 + (i>>3)*16 + (l>>4)*8 + (i&7)][nt*16 + (l&15)]
__global__ void cvt_frag_kernel(const float* __restrict__ src,
                                _Float16* __restrict__ dst,
                                int rows, int cols, int ktiles, int ntiles) {
  int total = ktiles * ntiles * 512;
  for (int o = blockIdx.x * blockDim.x + threadIdx.x; o < total;
       o += gridDim.x * blockDim.x) {
    int i    = o & 15;
    int lane = (o >> 4) & 31;
    int tile = o >> 9;
    int nt   = tile % ntiles;
    int kt   = tile / ntiles;
    int n    = nt * 16 + (lane & 15);
    int k    = kt * 32 + (i >> 3) * 16 + (lane >> 4) * 8 + (i & 7);
    float v  = (k < rows && n < cols) ? src[k * cols + n] : 0.0f;
    dst[o] = (_Float16)v;
  }
}

__global__ void zero_kernel(float* __restrict__ p, int n) {
  for (int i = blockIdx.x * blockDim.x + threadIdx.x; i < n;
       i += gridDim.x * blockDim.x)
    p[i] = 0.0f;
}

extern "C" void kernel_launch(void* const* d_in, const int* in_sizes, int n_in,
                              void* d_out, int out_size, void* d_ws, size_t ws_size,
                              hipStream_t stream) {
  (void)in_sizes; (void)n_in; (void)out_size; (void)ws_size;
  constexpr int N_PC = 60000, N_GRID = 15000;
  constexpr int E_PC2G = 300000, E_G2G = 300000, E_G2PC = 300000;

  const int*   pc2g_idx = (const int*)d_in[0];
  const float* pc2g_ef  = (const float*)d_in[1];
  const float* pc2g_nf  = (const float*)d_in[2];
  const int*   g2g_idx  = (const int*)d_in[3];
  const float* g2g_ef   = (const float*)d_in[4];
  const int*   g2pc_idx = (const int*)d_in[5];
  const float* g2pc_ef  = (const float*)d_in[6];

  struct FfnP {
    const float *W1, *W2, *b1, *b2, *beta, *g;
    int din, dout, kt1, nt2;
    _Float16 *w1f, *w2f;
  };
  // sorted outer keys: (ln, din, dout)
  static const struct { int ln, din, dout; } spec[19] = {
    {1,   4, 64},  // 0  dec_ee
    {0,  64,  4},  // 1  dec_no
    {1, 192, 64},  // 2  dec_pe
    {1, 128, 64},  // 3  dec_pn
    {1,   4, 64},  // 4  enc_ee
    {1,   8, 64},  // 5  enc_en
    {0,  64, 64},  // 6  enc_no
    {1, 192, 64},  // 7  enc_pe
    {1, 128, 64},  // 8  enc_pn
    {1, 192, 64},  // 9  proc_e0
    {1, 192, 64},  // 10 proc_e1
    {1, 192, 64},  // 11 proc_e2
    {1, 192, 64},  // 12 proc_e3
    {1,   4, 64},  // 13 proc_ee
    {1, 128, 64},  // 14 proc_n0
    {1, 128, 64},  // 15 proc_n1
    {1, 128, 64},  // 16 proc_n2
    {1, 128, 64},  // 17 proc_n3
    {0,  64, 64},  // 18 proc_no
  };
  enum { DEC_EE = 0, DEC_NO, DEC_PE, DEC_PN, ENC_EE, ENC_EN, ENC_NO, ENC_PE,
         ENC_PN, PROC_E0, PROC_E1, PROC_E2, PROC_E3, PROC_EE,
         PROC_N0, PROC_N1, PROC_N2, PROC_N3, PROC_NO };

  FfnP F[19];
  int p = 7;
  for (int i = 0; i < 19; ++i) {
    F[i].W1 = (const float*)d_in[p++];
    F[i].W2 = (const float*)d_in[p++];
    F[i].b1 = (const float*)d_in[p++];
    F[i].b2 = (const float*)d_in[p++];
    if (spec[i].ln) { F[i].beta = (const float*)d_in[p++];
                      F[i].g    = (const float*)d_in[p++]; }
    else            { F[i].beta = nullptr; F[i].g = nullptr; }
    F[i].din = spec[i].din;  F[i].dout = spec[i].dout;
    F[i].kt1 = (spec[i].din + 31) / 32;
    F[i].nt2 = (spec[i].dout + 15) / 16;
  }

  // ---- carve workspace ----
  char* ws = (char*)d_ws;
  size_t off = 0;
  auto carve = [&](size_t bytes) -> void* {
    off = (off + 255) & ~(size_t)255;
    void* r = ws + off;
    off += bytes;
    return r;
  };
  for (int i = 0; i < 19; ++i) {
    F[i].w1f = (_Float16*)carve((size_t)F[i].kt1 * 4 * 512 * 2);
    F[i].w2f = (_Float16*)carve((size_t)2 * F[i].nt2 * 512 * 2);
  }
  float* nBuf    = (float*)carve((size_t)N_PC   * 64 * 4);  // pc node latents
  float* edgeBuf = (float*)carve((size_t)E_PC2G * 64 * 4);  // e / ef / ef2 / nn
  float* meBuf   = (float*)carve((size_t)N_PC   * 64 * 4);  // scatter sums
  float* cntBuf  = (float*)carve((size_t)N_PC        * 4);  // scatter counts
  float* nfBuf   = (float*)carve((size_t)N_GRID * 64 * 4);  // grid latents

  // ---- weight pre-swizzle (tiny launches, L2-resident) ----
  for (int i = 0; i < 19; ++i) {
    int tot1 = F[i].kt1 * 4 * 512;
    cvt_frag_kernel<<<(tot1 + 255) / 256, 256, 0, stream>>>(
        F[i].W1, F[i].w1f, F[i].din, 64, F[i].kt1, 4);
    int tot2 = 2 * F[i].nt2 * 512;
    cvt_frag_kernel<<<(tot2 + 255) / 256, 256, 0, stream>>>(
        F[i].W2, F[i].w2f, 64, F[i].dout, 2, F[i].nt2);
  }

  auto ffn = [&](int fi,
                 const float* sA, const int* iA, int dA,
                 const float* sB, const int* iB, const float* cB, int dB,
                 const float* sC, int dC,
                 int actv, const float* res,
                 float* outp, int ldo,
                 float* efp, float* ssum, float* scnt, const int* sidx,
                 int nrows) {
    const FfnP& f = F[fi];
    int blocks = ((nrows + TR - 1) / TR + WPB - 1) / WPB;
    ffn_wmma_kernel<<<blocks, WPB * 32, 0, stream>>>(
        sA, iA, dA, sB, iB, cB, dB, sC, dC,
        f.w1f, f.kt1, f.b1, f.w2f, f.nt2, f.b2, f.dout,
        f.g, f.beta, actv, res, outp, ldo, efp, ssum, scnt, sidx, nrows);
  };
  auto zero = [&](float* ptr, int n) {
    zero_kernel<<<1024, 256, 0, stream>>>(ptr, n);
  };

  // ================= encoder =================
  // n = FFN_enc_en(pc node feats)      (60000 x 8 -> 64, LN)
  ffn(ENC_EN, pc2g_nf, nullptr, 8, nullptr, nullptr, nullptr, 0, nullptr, 0,
      0, nullptr, nBuf, 64, nullptr, nullptr, nullptr, nullptr, N_PC);
  // e = FFN_enc_ee(pc2g edge feats)    (300000 x 4 -> 64, LN) -> edgeBuf
  ffn(ENC_EE, pc2g_ef, nullptr, 4, nullptr, nullptr, nullptr, 0, nullptr, 0,
      0, nullptr, edgeBuf, 64, nullptr, nullptr, nullptr, nullptr, E_PC2G);
  // ne = FFN_enc_pe(n[s], n[r], e); scatter-mean into me over grid
  zero(meBuf, N_GRID * 64); zero(cntBuf, N_GRID);
  ffn(ENC_PE, nBuf, pc2g_idx, 64, nBuf, pc2g_idx + 1, nullptr, 64, edgeBuf, 64,
      0, nullptr, nullptr, 0, nullptr, meBuf, cntBuf, pc2g_idx + 1, E_PC2G);
  // nl = FFN_enc_pn(n[:NG], me) + n[:NG] -> nf
  ffn(ENC_PN, nBuf, nullptr, 64, meBuf, nullptr, cntBuf, 64, nullptr, 0,
      0, nBuf, nfBuf, 64, nullptr, nullptr, nullptr, nullptr, N_GRID);
  // nf = FFN_enc_no(nf)   (no LN, in place)
  ffn(ENC_NO, nfBuf, nullptr, 64, nullptr, nullptr, nullptr, 0, nullptr, 0,
      0, nullptr, nfBuf, 64, nullptr, nullptr, nullptr, nullptr, N_GRID);

  // ================= processor =================
  // ef = FFN_proc_ee(g2g edge feats) -> edgeBuf
  ffn(PROC_EE, g2g_ef, nullptr, 4, nullptr, nullptr, nullptr, 0, nullptr, 0,
      0, nullptr, edgeBuf, 64, nullptr, nullptr, nullptr, nullptr, E_G2G);
  for (int i = 0; i < 4; ++i) {
    zero(meBuf, N_GRID * 64); zero(cntBuf, N_GRID);
    // ne = FFN_proc_ei(nf[s], nf[r], ef); ef += ne; scatter-mean -> me
    ffn(PROC_E0 + i, nfBuf, g2g_idx, 64, nfBuf, g2g_idx + 1, nullptr, 64,
        edgeBuf, 64, 0, nullptr, nullptr, 0, edgeBuf,
        meBuf, cntBuf, g2g_idx + 1, E_G2G);
    // nf = nf + FFN_proc_ni(nf, me)
    ffn(PROC_N0 + i, nfBuf, nullptr, 64, meBuf, nullptr, cntBuf, 64, nullptr, 0,
        0, nfBuf, nfBuf, 64, nullptr, nullptr, nullptr, nullptr, N_GRID);
  }
  // nf = FFN_proc_no(nf)
  ffn(PROC_NO, nfBuf, nullptr, 64, nullptr, nullptr, nullptr, 0, nullptr, 0,
      0, nullptr, nfBuf, 64, nullptr, nullptr, nullptr, nullptr, N_GRID);

  // ================= decoder =================
  // ef2 = FFN_dec_ee(g2pc edge feats) -> edgeBuf
  ffn(DEC_EE, g2pc_ef, nullptr, 4, nullptr, nullptr, nullptr, 0, nullptr, 0,
      0, nullptr, edgeBuf, 64, nullptr, nullptr, nullptr, nullptr, E_G2PC);
  // ne = FFN_dec_pe(nf[s], n[r], ef2); scatter-mean over pc nodes
  zero(meBuf, N_PC * 64); zero(cntBuf, N_PC);
  ffn(DEC_PE, nfBuf, g2pc_idx, 64, nBuf, g2pc_idx + 1, nullptr, 64, edgeBuf, 64,
      0, nullptr, nullptr, 0, nullptr, meBuf, cntBuf, g2pc_idx + 1, E_G2PC);
  // nn = FFN_dec_pn(n, me) -> edgeBuf (reused, 60000 x 64)
  ffn(DEC_PN, nBuf, nullptr, 64, meBuf, nullptr, cntBuf, 64, nullptr, 0,
      0, nullptr, edgeBuf, 64, nullptr, nullptr, nullptr, nullptr, N_PC);
  // out = FFN_dec_no(nn)  (hidden act = sigmoid, dout = 4) -> d_out
  ffn(DEC_NO, edgeBuf, nullptr, 64, nullptr, nullptr, nullptr, 0, nullptr, 0,
      1, nullptr, (float*)d_out, 4, nullptr, nullptr, nullptr, nullptr, N_PC);
}
